// RecurTreeGen_31301721653838
// MI455X (gfx1250) — matrix operand order repaired and checked by reading, not compiled, not run on hardware
//
#include <hip/hip_runtime.h>
#include <hip/hip_bf16.h>

typedef __attribute__((ext_vector_type(16))) _Float16 v16h;
typedef __attribute__((ext_vector_type(8)))  float    v8f;
typedef __attribute__((ext_vector_type(4)))  float    f4;
typedef int iv4 __attribute__((vector_size(16)));     // 4 x i32 payload type

#define DIM   128      // embed dim
#define NCOLS 640      // 5 * DIM gate columns
#define KDIM  256      // 2 * DIM reduction
#define NT    40       // NCOLS / 16 column tiles
#define KS    8        // KDIM / 32 k-steps
#define CHUNK_F16 (5 * 8 * 32 * 16)   // one column-group B chunk: 40 KB as f16

#define AS1 __attribute__((address_space(1)))
#define AS3 __attribute__((address_space(3)))

#if __has_builtin(__builtin_amdgcn_global_load_async_to_lds_b128) && \
    __has_builtin(__builtin_amdgcn_s_wait_asynccnt)
#define USE_ASYNC_LDS 1
#else
#define USE_ASYNC_LDS 0
#endif

// ---------------------------------------------------------------------------
// Repack W [256][640] f32 -> B-panel f16 [NT][KS][32 lanes][16 f16]
// matching the wave32 WMMA B-fragment layout for v_wmma_f32_16x16x32_f16:
//   lane 0..15  : N = lane,    K = 0..7, 16..23  (2 f16 per VGPR, 8 VGPRs)
//   lane 16..31 : N = lane-16, K = 8..15, 24..31
// ---------------------------------------------------------------------------
__global__ void pack_w_kernel(const float* __restrict__ W, _Float16* __restrict__ Bp) {
    int idx  = blockIdx.x * 256 + threadIdx.x;   // one f16 output element each
    int j    = idx & 15;                         // K offset within lane's fragment
    int lane = (idx >> 4) & 31;
    int s    = (idx >> 9) & 7;                   // k-step
    int t    = idx >> 12;                        // n-tile 0..39
    int k = 32 * s + ((lane & 16) ? 8 : 0) + (j & 7) + ((j & 8) ? 16 : 0);
    int n = 16 * t + (lane & 15);
    Bp[idx] = (_Float16)W[k * NCOLS + n];
}

__device__ __forceinline__ float sigm(float x) {
    return 1.0f / (1.0f + __expf(-x));
}
__device__ __forceinline__ float fast_tanh(float x) {
    x = fminf(fmaxf(x, -15.0f), 15.0f);
    float e = __expf(2.0f * x);
    return (e - 1.0f) / (e + 1.0f);
}

// Stage column-group g's B chunk (5 gate tiles x 8 k-steps, 40 KB) into LDS.
// Async path: GLOBAL_LOAD_ASYNC_TO_LDS_B128 (ASYNCcnt), no VGPR round trip.
__device__ __forceinline__ void stage_chunk(const _Float16* __restrict__ Bp,
                                            _Float16* __restrict__ dst,
                                            int g, int tid) {
#pragma unroll
    for (int i = 0; i < 10; ++i) {
        int f    = i * 256 + tid;              // float4 index 0..2559
        int cell = f >> 1;                     // 32B cell = (q*8+s)*32 + lane'
        int half = f & 1;
        int q    = cell >> 8;
        int rem  = cell & 255;
        int s    = rem >> 5;
        int ln   = rem & 31;
        const _Float16* gp = Bp + ((((q * 8 + g) * 8 + s) * 32 + ln) * 16) + half * 8;
        _Float16*       lp = dst + f * 8;
#if USE_ASYNC_LDS
        __builtin_amdgcn_global_load_async_to_lds_b128(
            (AS1 iv4*)gp, (AS3 iv4*)lp, 0, 0);
#else
        *(f4*)lp = *(const f4*)gp;
#endif
    }
}

// ---------------------------------------------------------------------------
// Fused gather + GEMM(WMMA f16->f32) + TreeLSTM cell epilogue.
// 256 threads = 8 waves; wave w owns rows [strip, strip+16).
// B chunks double-buffered in LDS; async fill of chunk g+1 overlaps the
// 40 WMMAs + epilogue of chunk g.
// ---------------------------------------------------------------------------
__global__ __launch_bounds__(256, 1)
void treelstm_wmma_kernel(const float* __restrict__ h_bot, const float* __restrict__ c_bot,
                          const float* __restrict__ h_buf, const float* __restrict__ c_buf,
                          const int* __restrict__ bl, const int* __restrict__ pl,
                          const int* __restrict__ br, const int* __restrict__ pr,
                          const _Float16* __restrict__ Bp, const float* __restrict__ bias,
                          float* __restrict__ out, int Kn, int M)
{
    __shared__ __align__(16) _Float16 ldsB[2][CHUNK_F16];   // 2 x 40 KB

    const int tid  = threadIdx.x;
    const int lane = tid & 31;
    const int wave = tid >> 5;
    const int strip = (blockIdx.x * 8 + wave) * 16;

    // ---- per-lane A-row source pointers (h side) ----
    int arow = strip + (lane & 15);
    if (arow >= M) arow = 0;                       // clamp (M divisible in practice)
    const float* hl = (arow < Kn) ? h_bot + (size_t)bl[arow] * DIM
                                  : h_buf + (size_t)pl[arow - Kn] * DIM;
    const float* hr = (arow < Kn) ? h_bot + (size_t)br[arow] * DIM
                                  : h_buf + (size_t)pr[arow - Kn] * DIM;

    // ---- build the 8 A fragments (16x32 f16 each) for this 16-row strip ----
    // A layout per lane: elems 0..7  -> K = kb + koff + 0..7
    //                    elems 8..15 -> K = kb + 16 + koff + 0..7
    // koff = 0 for lanes 0..15, 8 for lanes 16..31.
    const int koff = (lane & 16) ? 8 : 0;
    v16h A[8];
#pragma unroll
    for (int s = 0; s < 8; ++s) {
        const float* src = (s < 4) ? hl : hr;      // concat [h_l ; h_r] along K
        int kb = (s & 3) * 32;
        f4 r0a = *(const f4*)(src + kb + koff);
        f4 r0b = *(const f4*)(src + kb + koff + 4);
        f4 r1a = *(const f4*)(src + kb + 16 + koff);
        f4 r1b = *(const f4*)(src + kb + 16 + koff + 4);
        v16h a;
        a[0]  = (_Float16)r0a.x;  a[1]  = (_Float16)r0a.y;
        a[2]  = (_Float16)r0a.z;  a[3]  = (_Float16)r0a.w;
        a[4]  = (_Float16)r0b.x;  a[5]  = (_Float16)r0b.y;
        a[6]  = (_Float16)r0b.z;  a[7]  = (_Float16)r0b.w;
        a[8]  = (_Float16)r1a.x;  a[9]  = (_Float16)r1a.y;
        a[10] = (_Float16)r1a.z;  a[11] = (_Float16)r1a.w;
        a[12] = (_Float16)r1b.x;  a[13] = (_Float16)r1b.y;
        a[14] = (_Float16)r1b.z;  a[15] = (_Float16)r1b.w;
        A[s] = a;
    }

    // prologue: start the first chunk's fill
    stage_chunk(Bp, &ldsB[0][0], 0, tid);

    // ---- loop over 8 column groups of 16 (covers d = 0..127 of each gate) ----
    for (int g = 0; g < 8; ++g) {
        const int buf = g & 1;
#if USE_ASYNC_LDS
        __builtin_amdgcn_s_wait_asynccnt(0);       // my async fills landed
#endif
        __syncthreads();                           // everyone's fills landed,
                                                   // and buf^1 reads (g-1) done
        if (g < 7) {
            stage_chunk(Bp, &ldsB[buf ^ 1][0], g + 1, tid);
            __builtin_prefetch(Bp + (size_t)(g + 1) * (8 * 32 * 16), 0, 0);
        }

        // 5 gate accumulators, 8 k-steps each -> 40 WMMAs per group
        const _Float16* lb = &ldsB[buf][0];
        v8f acc[5] = {};
#pragma unroll
        for (int q = 0; q < 5; ++q) {
#pragma unroll
            for (int s = 0; s < 8; ++s) {
                v16h bfrag = *(const v16h*)(lb + ((q * 8 + s) * 32 + lane) * 16);
                acc[q] = __builtin_amdgcn_wmma_f32_16x16x32_f16(
                    /*neg_a=*/false, A[s], /*neg_b=*/false, bfrag,
                    /*c_mod=*/(short)0, acc[q], /*reuse_a=*/false, /*reuse_b=*/false);
            }
        }

        // ---- fused TreeLSTM epilogue for columns d = 16g .. 16g+15 ----
        int col = g * 16 + (lane & 15);
        float b_i  = bias[col];
        float b_o  = bias[128 + col];
        float b_u  = bias[256 + col];
        float b_fl = bias[384 + col];
        float b_fr = bias[512 + col];
        int rowoff = (lane & 16) ? 8 : 0;
#pragma unroll
        for (int j = 0; j < 8; ++j) {
            int m = strip + rowoff + j;            // C layout: VGPR j -> row
            if (m < M) {
                const float* cl = (m < Kn) ? c_bot + (size_t)bl[m] * DIM
                                           : c_buf + (size_t)pl[m - Kn] * DIM;
                const float* cr = (m < Kn) ? c_bot + (size_t)br[m] * DIM
                                           : c_buf + (size_t)pr[m - Kn] * DIM;
                float clv = cl[col];
                float crv = cr[col];
                float gi  = acc[0][j] + b_i;
                float go  = acc[1][j] + b_o;
                float gu  = acc[2][j] + b_u;
                float gfl = acc[3][j] + b_fl;
                float gfr = acc[4][j] + b_fr;
                float c = sigm(gi) * fast_tanh(gu) + sigm(gfl) * clv + sigm(gfr) * crv;
                float h = sigm(go) * fast_tanh(c);
                out[(size_t)m * DIM + col]       = h;   // out[0] = h
                out[(size_t)(M + m) * DIM + col] = c;   // out[1] = c
            }
        }
    }
}

extern "C" void kernel_launch(void* const* d_in, const int* in_sizes, int n_in,
                              void* d_out, int out_size, void* d_ws, size_t ws_size,
                              hipStream_t stream) {
    const float* h_bot = (const float*)d_in[0];
    const float* c_bot = (const float*)d_in[1];
    const float* h_buf = (const float*)d_in[2];
    const float* c_buf = (const float*)d_in[3];
    const int*   bl    = (const int*)d_in[4];
    const int*   pl    = (const int*)d_in[5];
    const int*   br    = (const int*)d_in[6];
    const int*   pr    = (const int*)d_in[7];
    const float* W     = (const float*)d_in[8];
    const float* bias  = (const float*)d_in[9];

    const int Kn = in_sizes[4];        // 200000
    const int M  = 2 * Kn;             // 400000 merge targets

    _Float16* Bp = (_Float16*)d_ws;    // 40*8*32*16 f16 = 320 KB

    const int wElems  = in_sizes[8];   // 256*640 = 163840
    pack_w_kernel<<<wElems / 256, 256, 0, stream>>>(W, Bp);

    const int nwg = (M + 127) / 128;   // 8 waves * 16 rows per WG
    treelstm_wmma_kernel<<<nwg, 256, 0, stream>>>(
        h_bot, c_bot, h_buf, c_buf, bl, pl, br, pr, Bp, bias,
        (float*)d_out, Kn, M);
}
